// Recovery_89661737271702
// MI455X (gfx1250) — matrix operand order repaired and checked by reading, not compiled, not run on hardware
//
#include <hip/hip_runtime.h>
#include <hip/hip_bf16.h>

// ---------------------------------------------------------------------------
// 3-layer GRU (reset_after) + dense sigmoid head for MI455X (gfx1250).
// B=128, T=512, H=D=256, 3H=768, out=64.
//
//  - fp16 storage / fp32 accumulate via v_wmma_f32_16x16x32_f16.
//  - Per layer: parallel GEMM xW = x @ W + b0 (wave-per-16x16-tile), then a
//    single-WGP persistent scan kernel with double-buffered h in LDS.
//  - Weights pre-transposed to [N][K] f16 so B fragments are contiguous.
//  - A laundered zero *offset* (not pointer) keeps B-fragment loads inside the
//    t loop (no hoist->spill) while preserving global address-space provenance
//    (global_load_b128, not flat_load_b128).
//  - Gate math uses v_rcp_f32-based sigmoid/tanh (serial-path VALU matters).
// ---------------------------------------------------------------------------

typedef __attribute__((ext_vector_type(16))) _Float16 v16h;
typedef __attribute__((ext_vector_type(8)))  _Float16 v8h;
typedef __attribute__((ext_vector_type(8)))  float    v8f;

#define GRU_B   128
#define GRU_T   512
#define GRU_H   256
#define GRU_3H  768
#define GRU_BT  (GRU_B * GRU_T)   // 65536
#define LDH     264               // LDS row stride in halves (256 + 8 pad)

static __device__ __forceinline__ v8h ld8g(const _Float16* p) {
  return *reinterpret_cast<const v8h*>(p);
}
static __device__ __forceinline__ v16h ld16g(const _Float16* p) {
  return *reinterpret_cast<const v16h*>(p);
}
// fast sigmoid: v_exp + v_rcp (exact saturation at +/-inf)
static __device__ __forceinline__ float fast_sig(float x) {
  return __builtin_amdgcn_rcpf(1.0f + __expf(-x));
}
// fast tanh: tanh(x) = 2*sigmoid(2x) - 1
static __device__ __forceinline__ float fast_tanh(float x) {
  return 2.0f * __builtin_amdgcn_rcpf(1.0f + __expf(-2.0f * x)) - 1.0f;
}

// ---------------------------------------------------------------------------
// Elementwise: f32 -> f16 cast (activations)
// ---------------------------------------------------------------------------
__global__ void cast_f32_to_f16_kernel(const float* __restrict__ src,
                                       _Float16* __restrict__ dst, int total) {
  for (int i = blockIdx.x * blockDim.x + threadIdx.x; i < total;
       i += gridDim.x * blockDim.x) {
    dst[i] = (_Float16)src[i];
  }
}

// ---------------------------------------------------------------------------
// Transpose + cast weights: src [L][K][N] f32 -> dst [L][N][K] f16
// ---------------------------------------------------------------------------
__global__ void transpose_cast_kernel(const float* __restrict__ src,
                                      _Float16* __restrict__ dst,
                                      int K, int N, int total) {
  for (int i = blockIdx.x * blockDim.x + threadIdx.x; i < total;
       i += gridDim.x * blockDim.x) {
    int n = i % N;
    int rem = i / N;
    int k = rem % K;
    int l = rem / K;
    dst[((size_t)l * N + n) * K + k] = (_Float16)src[i];
  }
}

// ---------------------------------------------------------------------------
// Wave-per-tile WMMA GEMM:  C[M][Nout] = A[M][256] @ Bt[Nout][256]^T + bias
// EPI==0: plain bias add.  EPI==1: bias + sigmoid (dense head).
// ---------------------------------------------------------------------------
template <int EPI>
__global__ __launch_bounds__(128) void
wmma_gemm_kernel(const _Float16* __restrict__ A, const _Float16* __restrict__ Bt,
                 const float* __restrict__ bias, float* __restrict__ C,
                 int ntN, int n_tiles, int Nout) {
  const int wave = threadIdx.x >> 5;
  const int lane = threadIdx.x & 31;
  const int tile = blockIdx.x * 4 + wave;
  if (tile >= n_tiles) return;           // wave-uniform -> EXEC stays all-ones

  const int mt = tile / ntN;
  const int nt = tile % ntN;
  const int m0 = mt * 16;
  const int n0 = nt * 16;
  const int half = lane >> 4;            // 0 or 1 (half-wave)
  const int l16  = lane & 15;

  v8f acc = {};
  const _Float16* arow = A + (size_t)(m0 + l16) * GRU_H;      // A row, K-major
  const _Float16* bcol = Bt + (size_t)(n0 + l16) * GRU_H;     // B col, K-major

#pragma unroll
  for (int kk = 0; kk < 8; ++kk) {
    const int ka = kk * 32 + half * 16;   // A: lanes 0-15 K=0..15, 16-31 K=16..31
    v8h alo = ld8g(arow + ka);
    v8h ahi = ld8g(arow + ka + 8);        // A: VGPR4-7 K span (+8)
    v16h a = __builtin_shufflevector(alo, ahi, 0, 1, 2, 3, 4, 5, 6, 7,
                                     8, 9, 10, 11, 12, 13, 14, 15);
    const int kb = kk * 32 + half * 16;   // B: lanes 0-15 K=0..15, 16-31 K=16..31
    v16h b = ld16g(bcol + kb);
    acc = __builtin_amdgcn_wmma_f32_16x16x32_f16(false, a, false, b,
                                                 (short)0, acc, false, false);
  }

  const int n = n0 + l16;
  const float bn = bias[n];
#pragma unroll
  for (int i = 0; i < 8; ++i) {
    const int m = m0 + half * 8 + i;      // C layout: M = i + 8*(lane>=16)
    float v = acc[i] + bn;
    if (EPI == 1) v = fast_sig(v);
    C[(size_t)m * Nout + n] = v;
  }
}

// ---------------------------------------------------------------------------
// Persistent GRU scan for one layer. Single workgroup, 1024 threads (32 waves),
// h state double-buffered in LDS (2 x 128x264 f16 = 132KB). Each wave owns 4
// (z,r,h) tile-triples so the gate nonlinearity is wave-local; results are
// written straight into the other LDS buffer -> one barrier per step.
// ---------------------------------------------------------------------------
__global__ __launch_bounds__(1024) void
gru_scan_kernel(const float* __restrict__ xw,       // [BT][768], b0 included
                const _Float16* __restrict__ Ut,    // [768][256]
                const float* __restrict__ b1,       // [768] recurrent bias
                _Float16* __restrict__ x_next) {    // [B][T][H] f16
  __shared__ _Float16 hbuf[2][GRU_B * LDH];

  const int lane = threadIdx.x & 31;
  const int wave = threadIdx.x >> 5;     // 0..31
  const int half = lane >> 4;
  const int l16  = lane & 15;

  for (int i = threadIdx.x; i < GRU_B * LDH; i += 1024)
    hbuf[0][i] = (_Float16)0.0f;

  // hoist recurrent biases for this wave's 4 (mt,nt) triples into registers
  float bzq[4], brq[4], bhq[4];
#pragma unroll
  for (int q = 0; q < 4; ++q) {
    const int col = (((wave * 4 + q) & 15) * 16) + l16;
    bzq[q] = b1[col];
    brq[q] = b1[256 + col];
    bhq[q] = b1[512 + col];
  }
  __syncthreads();

  for (int t = 0; t < GRU_T; ++t) {
    const int cur = t & 1;
    const int nxt = cur ^ 1;

    // Launder a zero offset: keeps B-fragment loads un-hoistable across t
    // while preserving Ut's global provenance (-> global_load_b128).
    unsigned zoff = 0;
    asm volatile("" : "+s"(zoff));
    const _Float16* ut = Ut + zoff;

#pragma unroll 1
    for (int q = 0; q < 4; ++q) {
      const int idx = wave * 4 + q;      // 0..127 unique (mt,nt) triples
      const int mt = idx >> 4;           // 0..7  (batch tiles)
      const int nt = idx & 15;           // 0..15 (hidden-col tiles)
      const int m0 = mt * 16;
      const int n0 = nt * 16;

      v8f accz = {}, accr = {}, acch = {};
      const _Float16* hrow = &hbuf[cur][(m0 + l16) * LDH];
      const _Float16* bcol = ut + (size_t)(n0 + l16) * GRU_H;

#pragma unroll 1
      for (int kk = 0; kk < 8; ++kk) {
        const int ka = kk * 32 + half * 16;
        v8h alo = *reinterpret_cast<const v8h*>(hrow + ka);
        v8h ahi = *reinterpret_cast<const v8h*>(hrow + ka + 8);
        v16h a = __builtin_shufflevector(alo, ahi, 0, 1, 2, 3, 4, 5, 6, 7,
                                         8, 9, 10, 11, 12, 13, 14, 15);
        const int kb = kk * 32 + half * 16;
        v16h bz = ld16g(bcol + kb);                        // z block
        v16h br = ld16g(bcol + kb + 256 * GRU_H);          // r block
        v16h bh = ld16g(bcol + kb + 512 * GRU_H);          // h block
        accz = __builtin_amdgcn_wmma_f32_16x16x32_f16(false, a, false, bz,
                                                      (short)0, accz, false, false);
        accr = __builtin_amdgcn_wmma_f32_16x16x32_f16(false, a, false, br,
                                                      (short)0, accr, false, false);
        acch = __builtin_amdgcn_wmma_f32_16x16x32_f16(false, a, false, bh,
                                                      (short)0, acch, false, false);
      }

      // gate nonlinearity — all three gates for (b,col) live in this wave
      const int col = n0 + l16;
#pragma unroll
      for (int i = 0; i < 8; ++i) {
        const int brow = m0 + half * 8 + i;                 // batch index
        const size_t xrow = ((size_t)brow * GRU_T + t) * GRU_3H;
        const float xz = xw[xrow + col];
        const float xr = xw[xrow + 256 + col];
        const float xh = xw[xrow + 512 + col];
        if (i == 0 && t + 1 < GRU_T)                        // next-step prefetch
          __builtin_prefetch(&xw[xrow + GRU_3H + col], 0, 3);
        const float z  = fast_sig(xz + accz[i] + bzq[q]);
        const float r  = fast_sig(xr + accr[i] + brq[q]);
        const float hc = fast_tanh(xh + r * (acch[i] + bhq[q]));
        const float hp = (float)hbuf[cur][brow * LDH + col];
        const float hn = z * hp + (1.0f - z) * hc;
        const _Float16 hv = (_Float16)hn;
        hbuf[nxt][brow * LDH + col] = hv;                   // other buffer: safe
        x_next[((size_t)brow * GRU_T + t) * GRU_H + col] = hv;
      }
    }

    __syncthreads();   // h_t fully written / h_{t-1} reads done
  }
}

// ---------------------------------------------------------------------------
// Host-side orchestration
// ---------------------------------------------------------------------------
static inline size_t align256(size_t x) { return (x + 255) & ~(size_t)255; }

extern "C" void kernel_launch(void* const* d_in, const int* in_sizes, int n_in,
                              void* d_out, int out_size, void* d_ws, size_t ws_size,
                              hipStream_t stream) {
  (void)in_sizes; (void)n_in; (void)out_size; (void)ws_size;

  const float* h_in = (const float*)d_in[0];  // [B,T,256]
  const float* W    = (const float*)d_in[1];  // [3,256,768]
  const float* U    = (const float*)d_in[2];  // [3,256,768]
  const float* b    = (const float*)d_in[3];  // [3,2,768]
  const float* Wd   = (const float*)d_in[4];  // [256,64]
  const float* bd   = (const float*)d_in[5];  // [64]
  float* out = (float*)d_out;                 // [B,T,64]

  // workspace carve-up (f16 weights transposed, f16 activation ping-pong, f32 xW)
  char* ws = (char*)d_ws;
  size_t off = 0;
  _Float16* Wt  = (_Float16*)(ws + off); off = align256(off + (size_t)3 * GRU_3H * GRU_H * 2);
  _Float16* Ut  = (_Float16*)(ws + off); off = align256(off + (size_t)3 * GRU_3H * GRU_H * 2);
  _Float16* Wdt = (_Float16*)(ws + off); off = align256(off + (size_t)64 * GRU_H * 2);
  _Float16* xA  = (_Float16*)(ws + off); off = align256(off + (size_t)GRU_BT * GRU_H * 2);
  _Float16* xB  = (_Float16*)(ws + off); off = align256(off + (size_t)GRU_BT * GRU_H * 2);
  float*    xw  = (float*)   (ws + off); off = align256(off + (size_t)GRU_BT * GRU_3H * 4);

  // ---- weight conversion (transpose to [N][K] f16) ----
  {
    int total = 3 * GRU_H * GRU_3H;  // 589824
    transpose_cast_kernel<<<(total + 255) / 256, 256, 0, stream>>>(W, Wt, GRU_H, GRU_3H, total);
    transpose_cast_kernel<<<(total + 255) / 256, 256, 0, stream>>>(U, Ut, GRU_H, GRU_3H, total);
    int totd = GRU_H * 64;
    transpose_cast_kernel<<<(totd + 255) / 256, 256, 0, stream>>>(Wd, Wdt, GRU_H, 64, totd);
  }
  // ---- input activations -> f16 ----
  {
    int total = GRU_BT * GRU_H;      // 16.7M
    cast_f32_to_f16_kernel<<<16384, 256, 0, stream>>>(h_in, xA, total);
  }

  _Float16* x_cur = xA;
  _Float16* x_nxt = xB;

  for (int l = 0; l < 3; ++l) {
    const _Float16* Wtl = Wt + (size_t)l * GRU_3H * GRU_H;
    const _Float16* Utl = Ut + (size_t)l * GRU_3H * GRU_H;
    const float* b0 = b + (size_t)l * 2 * GRU_3H;
    const float* b1 = b0 + GRU_3H;

    // xW = x @ W + b0  : [65536 x 768], wave-per-tile
    {
      const int ntN = GRU_3H / 16;                   // 48
      const int n_tiles = (GRU_BT / 16) * ntN;       // 196608
      wmma_gemm_kernel<0><<<n_tiles / 4, 128, 0, stream>>>(
          x_cur, Wtl, b0, xw, ntN, n_tiles, GRU_3H);
    }
    // sequential scan on one WGP, h double-buffered in LDS
    gru_scan_kernel<<<1, 1024, 0, stream>>>(xw, Utl, b1, x_nxt);

    _Float16* tmp = x_cur; x_cur = x_nxt; x_nxt = tmp;
  }

  // dense head: sigmoid(y3 @ Wd + bd) -> [65536 x 64] f32
  {
    const int ntN = 64 / 16;                          // 4
    const int n_tiles = (GRU_BT / 16) * ntN;          // 16384
    wmma_gemm_kernel<1><<<n_tiles / 4, 128, 0, stream>>>(
        x_cur, Wdt, bd, out, ntN, n_tiles, 64);
  }
}
